// GraphUNet_1176821039773
// MI455X (gfx1250) — matrix operand order repaired and checked by reading, not compiled, not run on hardware
//
#include <hip/hip_runtime.h>
#include <hip/hip_bf16.h>
#include <math.h>

#define HID 128

typedef __bf16 bf16_t;
typedef bf16_t v16bf __attribute__((ext_vector_type(16)));
typedef float  v8f   __attribute__((ext_vector_type(8)));

// ---------------- device helpers ----------------
static __device__ __forceinline__ unsigned short f2bf(float f) {
  // round-to-nearest-even f32 -> bf16
  unsigned u = __float_as_uint(f);
  u += 0x7FFFu + ((u >> 16) & 1u);
  return (unsigned short)(u >> 16);
}
// order-preserving float<->uint encoding for atomicMax on floats
static __device__ __forceinline__ unsigned fenc(float f) {
  unsigned u = __float_as_uint(f);
  return (u & 0x80000000u) ? ~u : (u | 0x80000000u);
}
static __device__ __forceinline__ float fdec(unsigned e) {
  unsigned u = (e & 0x80000000u) ? (e & 0x7FFFFFFFu) : ~e;
  return __uint_as_float(u);
}

// ---------------- bf16 pre-conversion ----------------
// row-major f32 -> row-major bf16 (activations; converted ONCE per hetero call,
// amortized over the 6 GEMMs that read it; halves GEMM read traffic)
__global__ void k_cvt_bf16(const float* __restrict__ src, unsigned short* __restrict__ dst,
                           long long n) {
  long long t = (long long)blockIdx.x * blockDim.x + threadIdx.x;
  if (t < n) dst[t] = f2bf(src[t]);
}
// W[K,128] f32 -> Wt[128,K] bf16 (transposed so a lane's 16-K B-fragment run is
// contiguous -> two global_load_b128 instead of 16 strided b32)
__global__ void k_cvt_wt(const float* __restrict__ W, unsigned short* __restrict__ Wt,
                         int K) {
  int t = blockIdx.x * blockDim.x + threadIdx.x;
  if (t >= K * HID) return;
  int col = t / K, k = t % K;          // write coalesced in k
  Wt[t] = f2bf(W[(size_t)k * HID + col]);
}

// ---------------- WMMA GEMM: Y[R,128] = A[R,K] @ W[K,128] + bias ----------------
// A: row-major bf16 [R,K]; Wt: col-major bf16 [128,K]. One wave = 16x16 tile,
// 8 waves/block = 16 rows x 128 cols. Fully unrolled K loop (K=128 or 256):
// per step 4x b128 loads + v_wmma_f32_16x16x32_bf16.
template <int K>
__global__ __launch_bounds__(256) void k_gemm_bias_wmma(
    const unsigned short* __restrict__ A, const unsigned short* __restrict__ Wt,
    const float* __restrict__ bias, float* __restrict__ Y, int R) {
  const int lane = threadIdx.x & 31;
  const int wave = threadIdx.x >> 5;
  const int l16  = lane & 15;
  const int hi   = lane >> 4;          // 0 or 1 (half-wave)
  const int row0 = blockIdx.x << 4;
  const int col  = (wave << 4) + l16;  // 0..127
  int row = row0 + l16;
  if (row >= R) row = R - 1;           // clamp (store masked later); keeps EXEC full
  const unsigned short* arow = A + (size_t)row * K + (hi << 3);
  const unsigned short* bcol = Wt + (size_t)col * K + (hi << 3);
  v8f acc = {};
  union F16x16 { v16bf v; uint4 q[2]; };
#pragma unroll
  for (int k0 = 0; k0 < K; k0 += 32) {
    F16x16 a, b;
    // A frag: lanes0-15 K=[k0..k0+7, k0+16..k0+23]; lanes16-31 +8
    a.q[0] = *(const uint4*)(arow + k0);
    a.q[1] = *(const uint4*)(arow + k0 + 16);
    // B frag: lane holds column `col`, K=[k0..k0+7, k0+16..k0+23] (+8 for hi half)
    b.q[0] = *(const uint4*)(bcol + k0);
    b.q[1] = *(const uint4*)(bcol + k0 + 16);
    acc = __builtin_amdgcn_wmma_f32_16x16x32_bf16(false, a.v, false, b.v,
                                                  (short)0, acc, false, false);
  }
  const float bj = bias[col];
  float* yp = Y + (size_t)(row0 + (hi << 3)) * HID + col;
  if (row0 + 16 <= R) {
    // fast path (all but the single last row-block): no exec masking
#pragma unroll
    for (int v = 0; v < 8; ++v) yp[(size_t)v * HID] = acc[v] + bj;
  } else {
#pragma unroll
    for (int v = 0; v < 8; ++v) {
      int r = row0 + v + (hi << 3);    // D: VGPR v -> row v (+8 for hi half)
      if (r < R) yp[(size_t)v * HID] = acc[v] + bj;
    }
  }
}

// ---------------- elementwise / fill ----------------
__global__ void k_fill_f32(float* __restrict__ p, float v, long long n) {
  long long t = (long long)blockIdx.x * blockDim.x + threadIdx.x;
  if (t < n) p[t] = v;
}
__global__ void k_fill_u32(unsigned* __restrict__ p, unsigned v, long long n) {
  long long t = (long long)blockIdx.x * blockDim.x + threadIdx.x;
  if (t < n) p[t] = v;
}
__global__ void k_fill_i32(int* __restrict__ p, int v, long long n) {
  long long t = (long long)blockIdx.x * blockDim.x + threadIdx.x;
  if (t < n) p[t] = v;
}

// node encoder: h[R0,128], rows [0,N) = x@W+b, dummy row N = 0
__global__ void k_node_enc(const float* __restrict__ x, const float* __restrict__ W,
                           const float* __restrict__ b, float* __restrict__ h, int N) {
  long long t = (long long)blockIdx.x * blockDim.x + threadIdx.x;
  long long tot = (long long)(N + 1) * HID;
  if (t >= tot) return;
  int i = (int)(t >> 7), j = (int)(t & 127);
  float v = 0.f;
  if (i < N) {
    const float* xi = x + (size_t)i * 4;
    v = b[j] + xi[0] * W[j] + xi[1] * W[HID + j] + xi[2] * W[2 * HID + j] +
        xi[3] * W[3 * HID + j];
  }
  h[t] = v;
}

// hetero aggregation buffer init: sum of the three per-type output biases
__global__ void k_bias3_init(float* __restrict__ out, const float* __restrict__ b0,
                             const float* __restrict__ b1, const float* __restrict__ b2,
                             int R) {
  long long t = (long long)blockIdx.x * blockDim.x + threadIdx.x;
  if (t >= (long long)R * HID) return;
  int j = (int)(t & 127);
  out[t] = b0[j] + b1[j] + b2[j];
}

// per-(edge,head) attention logit + segment max (ordered-uint atomicMax)
__global__ void k_edge_logit(const int* __restrict__ src, const int* __restrict__ dst,
                             const float* __restrict__ ea, const float* __restrict__ We,
                             const float* __restrict__ att, const float* __restrict__ xl,
                             const float* __restrict__ xr, float* __restrict__ elog,
                             unsigned* __restrict__ nmax, int E) {
  int t = blockIdx.x * blockDim.x + threadIdx.x;
  if (t >= E * 4) return;
  int e = t >> 2, h = t & 3;
  int s = src[e], d = dst[e];
  const float* pl = xl + (size_t)s * HID + h * 32;
  const float* pr = xr + (size_t)d * HID + h * 32;
  const float* pa = att + h * 32;
  const float* pw = We ? (We + h * 32) : nullptr;
  float ew = ea ? ea[e] : 0.f;
  float acc = 0.f;
#pragma unroll 8
  for (int c = 0; c < 32; ++c) {
    float m = pl[c] + pr[c];
    if (pw) m += ew * pw[c];
    float lr = (m > 0.f) ? m : 0.2f * m;   // leaky_relu(0.2)
    acc += lr * pa[c];
  }
  elog[t] = acc;
  atomicMax(&nmax[(size_t)d * 4 + h], fenc(acc));
}

// ex = exp(logit - max[dst]); segment sum of ex
__global__ void k_edge_exp(const int* __restrict__ dst, const unsigned* __restrict__ nmax,
                           float* __restrict__ elog, float* __restrict__ nden, int E) {
  int t = blockIdx.x * blockDim.x + threadIdx.x;
  if (t >= E * 4) return;
  int e = t >> 2, h = t & 3;
  int d = dst[e];
  float ex = expf(elog[t] - fdec(nmax[(size_t)d * 4 + h]));
  elog[t] = ex;
  atomicAdd(&nden[(size_t)d * 4 + h], ex);
}

// out[dst,:] += xl[src,:] * alpha  (one thread per (edge, channel))
__global__ void k_edge_scatter(const int* __restrict__ src, const int* __restrict__ dst,
                               const float* __restrict__ xl, const float* __restrict__ elog,
                               const float* __restrict__ nden, float* __restrict__ out,
                               long long E) {
  long long t = (long long)blockIdx.x * blockDim.x + threadIdx.x;
  if (t >= E * HID) return;
  int e = (int)(t >> 7), j = (int)(t & 127), h = j >> 5;
  int s = src[e], d = dst[e];
  float alpha = elog[(size_t)e * 4 + h] / nden[(size_t)d * 4 + h];
  atomicAdd(&out[(size_t)d * HID + j], xl[(size_t)s * HID + j] * alpha);
}

__global__ void k_elu(const float* __restrict__ in, float* __restrict__ out, long long n) {
  long long t = (long long)blockIdx.x * blockDim.x + threadIdx.x;
  if (t >= n) return;
  float v = in[t];
  out[t] = (v > 0.f) ? v : expm1f(v);
}

// TopK score: tanh((x.w)/||w||)
__global__ void k_score(const float* __restrict__ xo, const float* __restrict__ w,
                        float* __restrict__ sc, int n) {
  int i = blockIdx.x * blockDim.x + threadIdx.x;
  if (i >= n) return;
  const float* xi = xo + (size_t)i * HID;
  float dot = 0.f, nn = 0.f;
#pragma unroll 8
  for (int j = 0; j < HID; ++j) { float wj = w[j]; dot += xi[j] * wj; nn += wj * wj; }
  sc[i] = tanhf(dot * rsqrtf(nn));
}

// stable descending rank (ties broken by index) -> perm/topv of top-k
__global__ __launch_bounds__(256) void k_rank_select(const float* __restrict__ s, int n,
                                                     int k, int* __restrict__ perm,
                                                     float* __restrict__ topv) {
  __shared__ float tile[256];
  int i = blockIdx.x * 256 + threadIdx.x;
  float si = (i < n) ? s[i] : 0.f;
  int r = 0;
  for (int base = 0; base < n; base += 256) {
    int j = base + (int)threadIdx.x;
    tile[threadIdx.x] = (j < n) ? s[j] : 0.f;
    __syncthreads();
    int lim = (n - base < 256) ? (n - base) : 256;
    for (int q = 0; q < lim; ++q) {
      float sj = tile[q];
      int jj = base + q;
      r += (int)((sj > si) || (sj == si && jj < i));
    }
    __syncthreads();
  }
  if (i < n && r < k) { perm[r] = i; topv[r] = si; }
}

__global__ void k_mp_scatter(const int* __restrict__ perm, int* __restrict__ mp, int k) {
  int r = blockIdx.x * blockDim.x + threadIdx.x;
  if (r < k) mp[perm[r]] = r;
}

// relabel edges; invalid edges routed to dummy node k
__global__ void k_remap(const int* __restrict__ s, const int* __restrict__ d,
                        const int* __restrict__ mp, int* __restrict__ ns,
                        int* __restrict__ nd, int k, int E) {
  int e = blockIdx.x * blockDim.x + threadIdx.x;
  if (e >= E) return;
  int ms = mp[s[e]], md = mp[d[e]];
  bool valid = (ms >= 0) && (md >= 0);
  ns[e] = valid ? ms : k;
  nd[e] = valid ? md : k;
}

// cur[r,:] = xo[perm[r],:]*topv[r]; dummy row k zeroed
__global__ void k_pool_gather(const float* __restrict__ xo, const int* __restrict__ perm,
                              const float* __restrict__ topv, float* __restrict__ cur, int k) {
  long long t = (long long)blockIdx.x * blockDim.x + threadIdx.x;
  if (t >= (long long)(k + 1) * HID) return;
  int r = (int)(t >> 7), j = (int)(t & 127);
  cur[t] = (r < k) ? xo[(size_t)perm[r] * HID + j] * topv[r] : 0.f;
}

// cat[i,0:128]=0, cat[i,128:256]=xs[i,:]; dummy row ni fully zero
__global__ void k_cat_base(const float* __restrict__ xs, float* __restrict__ cat, int ni) {
  long long t = (long long)blockIdx.x * blockDim.x + threadIdx.x;
  if (t >= (long long)(ni + 1) * 256) return;
  int i = (int)(t >> 8), j = (int)(t & 255);
  cat[t] = (i < ni && j >= 128) ? xs[(size_t)i * HID + (j - 128)] : 0.f;
}
// cat[perm[r],0:128] = xc[r,:]  (unpool scatter)
__global__ void k_cat_scatter(const float* __restrict__ xc, const int* __restrict__ perm,
                              float* __restrict__ cat, int k) {
  long long t = (long long)blockIdx.x * blockDim.x + threadIdx.x;
  if (t >= (long long)k * HID) return;
  int r = (int)(t >> 7), j = (int)(t & 127);
  cat[(size_t)perm[r] * 256 + j] = xc[t];
}

// final projection 128 -> 2
__global__ void k_out_proj(const float* __restrict__ xf, const float* __restrict__ W,
                           const float* __restrict__ b, float* __restrict__ out, int n) {
  int t = blockIdx.x * blockDim.x + threadIdx.x;
  if (t >= n * 2) return;
  int i = t >> 1, o = t & 1;
  const float* xi = xf + (size_t)i * HID;
  float acc = b[o];
#pragma unroll 8
  for (int j = 0; j < HID; ++j) acc += xi[j] * W[j * 2 + o];
  out[t] = acc;
}

// ---------------- host orchestration ----------------
struct GatP { const float *Wl, *bl, *Wr, *br, *att, *bias, *We; };
struct HetP { GatP g[3]; };  // 0=phys 1=align 2=logic
struct Edges { const int* s[3]; const int* d[3]; int E[3]; const float* ea[3]; };
struct Scratch {
  float *xl, *xr, *aggr, *elog, *nden; unsigned* nmax;
  unsigned short *abf, *wtb;   // bf16 activation copy + bf16 transposed weight
};

static inline int cdiv(long long a, long long b) { return (int)((a + b - 1) / b); }

static GatP take_gat(void* const* d_in, int& i, bool we) {
  GatP g;
  g.Wl  = (const float*)d_in[i++]; g.bl   = (const float*)d_in[i++];
  g.Wr  = (const float*)d_in[i++]; g.br   = (const float*)d_in[i++];
  g.att = (const float*)d_in[i++]; g.bias = (const float*)d_in[i++];
  g.We  = we ? (const float*)d_in[i++] : nullptr;
  return g;
}
static HetP take_het(void* const* d_in, int& i) {
  HetP h;
  h.g[0] = take_gat(d_in, i, true);
  h.g[1] = take_gat(d_in, i, false);
  h.g[2] = take_gat(d_in, i, true);
  return h;
}

static void launch_gemm(hipStream_t st, const unsigned short* abf, const float* W,
                        unsigned short* wtb, const float* bias, float* Y, int R, int K) {
  k_cvt_wt<<<cdiv((long long)K * HID, 256), 256, 0, st>>>(W, wtb, K);
  if (K == 128)
    k_gemm_bias_wmma<128><<<cdiv(R, 16), 256, 0, st>>>(abf, wtb, bias, Y, R);
  else
    k_gemm_bias_wmma<256><<<cdiv(R, 16), 256, 0, st>>>(abf, wtb, bias, Y, R);
}

static void run_gat(hipStream_t st, int n, int K, const int* src, const int* dst, int E,
                    const float* ea, const GatP& g, Scratch& S) {
  int R = n + 1;
  launch_gemm(st, S.abf, g.Wl, S.wtb, g.bl, S.xl, R, K);
  launch_gemm(st, S.abf, g.Wr, S.wtb, g.br, S.xr, R, K);
  k_fill_u32<<<cdiv((long long)R * 4, 256), 256, 0, st>>>(S.nmax, 0u, (long long)R * 4);
  k_fill_f32<<<cdiv((long long)R * 4, 256), 256, 0, st>>>(S.nden, 0.f, (long long)R * 4);
  k_edge_logit<<<cdiv((long long)E * 4, 256), 256, 0, st>>>(src, dst, ea, g.We, g.att,
                                                            S.xl, S.xr, S.elog, S.nmax, E);
  k_edge_exp<<<cdiv((long long)E * 4, 256), 256, 0, st>>>(dst, S.nmax, S.elog, S.nden, E);
  k_edge_scatter<<<cdiv((long long)E * HID, 256), 256, 0, st>>>(src, dst, S.xl, S.elog,
                                                                S.nden, S.aggr, E);
}

static void run_het(hipStream_t st, const float* xpad, int n, int K, const Edges& ed,
                    const HetP& hp, Scratch& S) {
  int R = n + 1;
  // convert activations to bf16 once; read by all 6 GEMMs of this hetero call
  k_cvt_bf16<<<cdiv((long long)R * K, 256), 256, 0, st>>>(xpad, S.abf, (long long)R * K);
  k_bias3_init<<<cdiv((long long)R * HID, 256), 256, 0, st>>>(
      S.aggr, hp.g[0].bias, hp.g[1].bias, hp.g[2].bias, R);
  for (int t = 0; t < 3; ++t)
    run_gat(st, n, K, ed.s[t], ed.d[t], ed.E[t], ed.ea[t], hp.g[t], S);
}

static void run_pool(hipStream_t st, const float* xo, const float* poolw, int n, int k,
                     const Edges& ein, Edges& eout, int* perm, float* topv, int* mp,
                     float* score, float* cur, int* es[3], int* ed_[3]) {
  k_score<<<cdiv(n, 256), 256, 0, st>>>(xo, poolw, score, n);
  k_rank_select<<<cdiv(n, 256), 256, 0, st>>>(score, n, k, perm, topv);
  k_fill_i32<<<cdiv(n, 256), 256, 0, st>>>(mp, -1, n);
  k_mp_scatter<<<cdiv(k, 256), 256, 0, st>>>(perm, mp, k);
  for (int t = 0; t < 3; ++t) {
    k_remap<<<cdiv(ein.E[t], 256), 256, 0, st>>>(ein.s[t], ein.d[t], mp, es[t], ed_[t],
                                                 k, ein.E[t]);
    eout.s[t] = es[t]; eout.d[t] = ed_[t]; eout.E[t] = ein.E[t]; eout.ea[t] = ein.ea[t];
  }
  k_pool_gather<<<cdiv((long long)(k + 1) * HID, 256), 256, 0, st>>>(xo, perm, topv, cur, k);
}

extern "C" void kernel_launch(void* const* d_in, const int* in_sizes, int n_in,
                              void* d_out, int out_size, void* d_ws, size_t ws_size,
                              hipStream_t stream) {
  (void)n_in; (void)out_size; (void)ws_size;
  const int N0 = in_sizes[0] / 4;          // 50000
  const int EP = in_sizes[1] / 2, EA = in_sizes[2] / 2, EL = in_sizes[3] / 2;
  const int N1 = (N0 + 1) / 2, N2 = (N1 + 1) / 2;
  const int R0 = N0 + 1, R1 = N1 + 1, R2 = N2 + 1;
  const int EMAX = (EP > EL ? EP : EL) > EA ? (EP > EL ? EP : EL) : EA;

  // -------- inputs --------
  const float* x = (const float*)d_in[0];
  const int* phys = (const int*)d_in[1];
  const int* alig = (const int*)d_in[2];
  const int* logi = (const int*)d_in[3];
  const float* phys_ea = (const float*)d_in[4];
  const float* logic_ea = (const float*)d_in[5];
  int pi = 6;
  const float* encW = (const float*)d_in[pi++];
  const float* encB = (const float*)d_in[pi++];
  HetP enc0 = take_het(d_in, pi);
  HetP enc1 = take_het(d_in, pi);
  const float* pool0 = (const float*)d_in[pi++];
  const float* pool1 = (const float*)d_in[pi++];
  HetP bott = take_het(d_in, pi);
  HetP dec0 = take_het(d_in, pi);
  HetP dec1 = take_het(d_in, pi);
  const float* outW = (const float*)d_in[pi++];
  const float* outB = (const float*)d_in[pi++];

  // -------- workspace partition --------
  char* w = (char*)d_ws;
  auto alloc = [&](size_t bytes) -> void* {
    void* p = (void*)w;
    w += (bytes + 255) & ~(size_t)255;
    return p;
  };
  float* hbuf  = (float*)alloc((size_t)R0 * HID * 4);   // level0 padded feats / final feats
  float* xl    = (float*)alloc((size_t)R0 * HID * 4);
  float* xr    = (float*)alloc((size_t)R0 * HID * 4);
  float* aggr  = (float*)alloc((size_t)R0 * HID * 4);
  float* xs0   = (float*)alloc((size_t)N0 * HID * 4);
  float* xs1   = (float*)alloc((size_t)N1 * HID * 4);
  float* cur1  = (float*)alloc((size_t)R1 * HID * 4);
  float* cur2  = (float*)alloc((size_t)R2 * HID * 4);
  float* cat   = (float*)alloc((size_t)R0 * 256 * 4);
  float* dcur1 = (float*)alloc((size_t)N1 * HID * 4);
  float* dcur2 = (float*)alloc((size_t)N2 * HID * 4);
  float* elog  = (float*)alloc((size_t)EMAX * 4 * 4);
  unsigned* nmax = (unsigned*)alloc((size_t)R0 * 4 * 4);
  float* nden  = (float*)alloc((size_t)R0 * 4 * 4);
  float* score = (float*)alloc((size_t)N0 * 4);
  int* perm0 = (int*)alloc((size_t)N1 * 4);
  float* topv0 = (float*)alloc((size_t)N1 * 4);
  int* perm1 = (int*)alloc((size_t)N2 * 4);
  float* topv1 = (float*)alloc((size_t)N2 * 4);
  int* mp = (int*)alloc((size_t)N0 * 4);
  unsigned short* abf = (unsigned short*)alloc((size_t)R0 * 256 * 2);  // bf16 activations
  unsigned short* wtb = (unsigned short*)alloc((size_t)256 * HID * 2); // bf16 W^T
  int* e1s[3], *e1d[3], *e2s[3], *e2d[3];
  const int Esz[3] = {EP, EA, EL};
  for (int t = 0; t < 3; ++t) {
    e1s[t] = (int*)alloc((size_t)Esz[t] * 4); e1d[t] = (int*)alloc((size_t)Esz[t] * 4);
    e2s[t] = (int*)alloc((size_t)Esz[t] * 4); e2d[t] = (int*)alloc((size_t)Esz[t] * 4);
  }

  Scratch S{xl, xr, aggr, elog, nden, nmax, abf, wtb};

  Edges L0, L1, L2;
  L0.s[0] = phys; L0.d[0] = phys + EP; L0.E[0] = EP; L0.ea[0] = phys_ea;
  L0.s[1] = alig; L0.d[1] = alig + EA; L0.E[1] = EA; L0.ea[1] = nullptr;
  L0.s[2] = logi; L0.d[2] = logi + EL; L0.E[2] = EL; L0.ea[2] = logic_ea;

  // 1) node encoder (padded with zero dummy row)
  k_node_enc<<<cdiv((long long)R0 * HID, 256), 256, 0, stream>>>(x, encW, encB, hbuf, N0);

  // 2) encoder level 0
  run_het(stream, hbuf, N0, HID, L0, enc0, S);
  k_elu<<<cdiv((long long)N0 * HID, 256), 256, 0, stream>>>(aggr, xs0, (long long)N0 * HID);
  // pool 0
  run_pool(stream, xs0, pool0, N0, N1, L0, L1, perm0, topv0, mp, score, cur1, e1s, e1d);

  // 3) encoder level 1
  run_het(stream, cur1, N1, HID, L1, enc1, S);
  k_elu<<<cdiv((long long)N1 * HID, 256), 256, 0, stream>>>(aggr, xs1, (long long)N1 * HID);
  // pool 1
  run_pool(stream, xs1, pool1, N1, N2, L1, L2, perm1, topv1, mp, score, cur2, e2s, e2d);

  // 4) bottleneck
  run_het(stream, cur2, N2, HID, L2, bott, S);
  k_elu<<<cdiv((long long)N2 * HID, 256), 256, 0, stream>>>(aggr, dcur2, (long long)N2 * HID);

  // 5) decoder i=1 (level-1 edges, in_dim=256)
  k_cat_base<<<cdiv((long long)R1 * 256, 256), 256, 0, stream>>>(xs1, cat, N1);
  k_cat_scatter<<<cdiv((long long)N2 * HID, 256), 256, 0, stream>>>(dcur2, perm1, cat, N2);
  run_het(stream, cat, N1, 256, L1, dec1, S);
  k_elu<<<cdiv((long long)N1 * HID, 256), 256, 0, stream>>>(aggr, dcur1, (long long)N1 * HID);

  // 6) decoder i=0 (level-0 edges, in_dim=256)
  k_cat_base<<<cdiv((long long)R0 * 256, 256), 256, 0, stream>>>(xs0, cat, N0);
  k_cat_scatter<<<cdiv((long long)N1 * HID, 256), 256, 0, stream>>>(dcur1, perm0, cat, N0);
  run_het(stream, cat, N0, 256, L0, dec0, S);
  k_elu<<<cdiv((long long)N0 * HID, 256), 256, 0, stream>>>(aggr, hbuf, (long long)N0 * HID);

  // 7) output projection
  k_out_proj<<<cdiv((long long)N0 * 2, 256), 256, 0, stream>>>(hbuf, outW, outB,
                                                               (float*)d_out, N0);
}
